// DeepLagrangianNetwork_35579509080614
// MI455X (gfx1250) — matrix operand order repaired and checked by reading, not compiled, not run on hardware
//
#include <hip/hip_runtime.h>
#include <math.h>

typedef __attribute__((ext_vector_type(16))) _Float16 v16h;
typedef __attribute__((ext_vector_type(8)))  _Float16 v8h;
typedef __attribute__((ext_vector_type(2)))  _Float16 v2h;
typedef __attribute__((ext_vector_type(8)))  float    v8f;
typedef __attribute__((ext_vector_type(2)))  float    v2f;

#define H2_STRIDE 68   // floats; 272B rows keep b128 alignment, conflict-free stores
#define W2_STRIDE 72   // halves; 144B rows, 16B aligned, near-conflict-free b128 loads

__global__ __launch_bounds__(256)
void dlan_fused_kernel(const float* __restrict__ x,
                       const float* __restrict__ W1, const float* __restrict__ b1,
                       const float* __restrict__ W2, const float* __restrict__ b2,
                       const float* __restrict__ WLd, const float* __restrict__ bLd,
                       const float* __restrict__ WLo, const float* __restrict__ bLo,
                       float* __restrict__ out,
                       int nrows, int numTiles, int nWaves) {
    __shared__ v2h    w1p[32 * 4];       // {W1[2p][d], W1[2p+1][d]} f16 pairs
    __shared__ v2h    b1p[32];           // {b1[2p], b1[2p+1]} f16 pairs
    __shared__ float  b2s[64];
    __shared__ float4 whv[64];           // {W_Ld[0][n], W_Ld[1][n], W_Lo[n], 0}
    __shared__ __align__(16) _Float16 w2h[64 * W2_STRIDE];  // W2 in f16, B-layout source
    __shared__ float  h2s[8 * 16 * H2_STRIDE];

    const int tid  = threadIdx.x;
    const int lane = tid & 31;
    const int lo   = lane & 15;          // row-within-tile / column-within-tile index
    const int hi   = (lane >> 4) & 1;    // half-wave select

    // ---- cooperative weight staging ----
    if (tid < 128) {                     // W1 as f16 pairs: thread -> pair p = tid>>2, dim d = tid&3
        const int p = tid >> 2;
        const int d = tid & 3;
        v2h w;
        w[0] = (_Float16)W1[(2 * p) * 4 + d];
        w[1] = (_Float16)W1[(2 * p + 1) * 4 + d];
        w1p[p * 4 + d] = w;
    }
    if (tid < 32) {
        v2h bb;
        bb[0] = (_Float16)b1[2 * tid];
        bb[1] = (_Float16)b1[2 * tid + 1];
        b1p[tid] = bb;
    }
    if (tid < 64) {
        b2s[tid] = b2[tid];
        whv[tid] = make_float4(WLd[tid], WLd[64 + tid], WLo[tid], 0.0f);
    }
    // convert W2 (64x64 f32, row-major) to f16 in LDS, row stride W2_STRIDE halves.
    {
        const int n  = tid >> 2;
        const int k0 = (tid & 3) * 16;
        const float4* src = reinterpret_cast<const float4*>(W2 + n * 64 + k0);
        float4 f0 = src[0], f1 = src[1], f2 = src[2], f3 = src[3];
        _Float16* dst = &w2h[n * W2_STRIDE + k0];
        dst[0]  = (_Float16)f0.x; dst[1]  = (_Float16)f0.y;
        dst[2]  = (_Float16)f0.z; dst[3]  = (_Float16)f0.w;
        dst[4]  = (_Float16)f1.x; dst[5]  = (_Float16)f1.y;
        dst[6]  = (_Float16)f1.z; dst[7]  = (_Float16)f1.w;
        dst[8]  = (_Float16)f2.x; dst[9]  = (_Float16)f2.y;
        dst[10] = (_Float16)f2.z; dst[11] = (_Float16)f2.w;
        dst[12] = (_Float16)f3.x; dst[13] = (_Float16)f3.y;
        dst[14] = (_Float16)f3.z; dst[15] = (_Float16)f3.w;
    }
    __syncthreads();

    const float bld0 = bLd[0];
    const float bld1 = bLd[1];
    const float blo  = bLo[0];

    const float4* x4   = reinterpret_cast<const float4*>(x);
    float4*       out4 = reinterpret_cast<float4*>(out);
    const int wbase = (tid >> 5) * (16 * H2_STRIDE);
    const int wid   = (blockIdx.x * blockDim.x + tid) >> 5;

    const v2h c001 = {(_Float16)0.01f, (_Float16)0.01f};

    for (int t = wid; t < numTiles; t += nWaves) {
        const int base = t * 16;
        const int row  = base + lo;
        const int qr   = row < nrows ? row : (nrows - 1);
        const float4 qv = x4[qr];

        v2h q2[4];
        {
            const _Float16 qx = (_Float16)qv.x, qy = (_Float16)qv.y;
            const _Float16 qz = (_Float16)qv.z, qw = (_Float16)qv.w;
            q2[0][0] = qx; q2[0][1] = qx;
            q2[1][0] = qy; q2[1][1] = qy;
            q2[2][0] = qz; q2[2][1] = qz;
            q2[3][0] = qw; q2[3][1] = qw;
        }

        // ---- layer 1: h1 = leaky(q @ W1^T + b1) in packed f16, built in A-fragment layout
        // lane l: row m = l&15; halves 0-7 <-> K=kb..kb+7, halves 8-15 <-> K=kb+16..kb+23,
        // kb = 32s + 8*hi; pair index pb = kb/2 = 16s + 4*hi
        v16h afr[2];
#pragma unroll
        for (int s = 0; s < 2; ++s) {
            const int pb = 16 * s + 4 * hi;
            v16h a;
#pragma unroll
            for (int p = 0; p < 4; ++p) {
                v2h acc0 = b1p[pb + p];          // k = kb+2p, kb+2p+1
                v2h acc1 = b1p[pb + 8 + p];      // k = kb+16+2p, kb+16+2p+1
#pragma unroll
                for (int d = 0; d < 4; ++d) {
                    acc0 += w1p[(pb + p) * 4 + d] * q2[d];
                    acc1 += w1p[(pb + 8 + p) * 4 + d] * q2[d];
                }
                acc0 = __builtin_elementwise_max(acc0, acc0 * c001);  // leaky relu
                acc1 = __builtin_elementwise_max(acc1, acc1 * c001);
                a[2 * p]     = acc0[0];
                a[2 * p + 1] = acc0[1];
                a[8 + 2 * p] = acc1[0];
                a[9 + 2 * p] = acc1[1];
            }
            afr[s] = a;
        }

        // ---- layer 2: h2 = h1 @ W2^T via 8x v_wmma_f32_16x16x32_f16, C starts at inline 0
        v8f cf[4];
#pragma unroll
        for (int j = 0; j < 4; ++j) {
            const v8h* bp = reinterpret_cast<const v8h*>(w2h)
                          + (16 * j + lo) * (W2_STRIDE / 8) + hi;
            const v8h b0a = bp[0], b0b = bp[2];   // s = 0
            const v8h b1a = bp[4], b1b = bp[6];   // s = 1
            const v16h bf0 = __builtin_shufflevector(b0a, b0b,
                0, 1, 2, 3, 4, 5, 6, 7, 8, 9, 10, 11, 12, 13, 14, 15);
            const v16h bf1 = __builtin_shufflevector(b1a, b1b,
                0, 1, 2, 3, 4, 5, 6, 7, 8, 9, 10, 11, 12, 13, 14, 15);
            v8f c{};
            c = __builtin_amdgcn_wmma_f32_16x16x32_f16(false, afr[0], false, bf0,
                                                       (short)0, c, false, false);
            c = __builtin_amdgcn_wmma_f32_16x16x32_f16(false, afr[1], false, bf1,
                                                       (short)0, c, false, false);
            cf[j] = c;
        }

        // ---- bias + leaky relu + spill h2 tile to wave-private LDS
        // C layout: element e of cf[j] is (m = e + 8*hi, n = 16*j + lo)
#pragma unroll
        for (int j = 0; j < 4; ++j) {
            const float bb = b2s[16 * j + lo];
#pragma unroll
            for (int e = 0; e < 8; ++e) {
                float v = cf[j][e] + bb;
                v = fmaxf(v, 0.01f * v);
                h2s[wbase + (e + 8 * hi) * H2_STRIDE + 16 * j + lo] = v;
            }
        }

        // ---- heads: row r = lo; this lane reduces n in [32*hi, 32*hi+32)
        // p01 packs the two Ld partial sums -> v_pk_fma_f32
        const int n0 = 32 * hi;
        const float* hrow = &h2s[wbase + lo * H2_STRIDE + n0];
        v2f p01 = {0.f, 0.f};
        float p2 = 0.f;
#pragma unroll
        for (int u = 0; u < 8; ++u) {
            const float4 h  = *reinterpret_cast<const float4*>(hrow + 4 * u);
            const float4 wa = whv[n0 + 4 * u + 0];
            const float4 wb = whv[n0 + 4 * u + 1];
            const float4 wc = whv[n0 + 4 * u + 2];
            const float4 wd = whv[n0 + 4 * u + 3];
            p01 += (v2f){h.x, h.x} * (v2f){wa.x, wa.y};
            p01 += (v2f){h.y, h.y} * (v2f){wb.x, wb.y};
            p01 += (v2f){h.z, h.z} * (v2f){wc.x, wc.y};
            p01 += (v2f){h.w, h.w} * (v2f){wd.x, wd.y};
            p2 = fmaf(h.x, wa.z, fmaf(h.y, wb.z, fmaf(h.z, wc.z, fmaf(h.w, wd.z, p2))));
        }
        float p0 = p01[0], p1 = p01[1];
        p0 += __shfl_xor(p0, 16, 32);
        p1 += __shfl_xor(p1, 16, 32);
        p2 += __shfl_xor(p2, 16, 32);

        // softplus (== jax.nn.softplus = max(x,0)+log1p(exp(-|x|))) and H = L L^T + 1e-9 I
        const float za = p0 + bld0;
        const float zb = p1 + bld1;
        const float da = fmaxf(za, 0.f) + log1pf(expf(-fabsf(za)));
        const float db = fmaxf(zb, 0.f) + log1pf(expf(-fabsf(zb)));
        const float dc = p2 + blo;
        const float od = da * dc;
        const float4 Hv = make_float4(fmaf(da, da, 1e-9f), od, od,
                                      fmaf(dc, dc, fmaf(db, db, 1e-9f)));
        if (lane < 16 && row < nrows) out4[row] = Hv;
    }
}

extern "C" void kernel_launch(void* const* d_in, const int* in_sizes, int n_in,
                              void* d_out, int out_size, void* d_ws, size_t ws_size,
                              hipStream_t stream) {
    (void)n_in; (void)out_size; (void)d_ws; (void)ws_size;
    const float* x   = (const float*)d_in[0];
    const float* W1  = (const float*)d_in[1];
    const float* b1  = (const float*)d_in[2];
    const float* W2  = (const float*)d_in[3];
    const float* b2  = (const float*)d_in[4];
    const float* WLd = (const float*)d_in[5];
    const float* bLd = (const float*)d_in[6];
    const float* WLo = (const float*)d_in[7];
    const float* bLo = (const float*)d_in[8];
    float* out = (float*)d_out;

    const int nrows    = in_sizes[0] / 4;
    const int numTiles = (nrows + 15) / 16;

    int blocks = 1024;                       // 8192 waves -> ~8 tiles/wave
    const int maxBlocks = (numTiles + 7) / 8;
    if (blocks > maxBlocks) blocks = maxBlocks;
    if (blocks < 1) blocks = 1;
    const int nWaves = blocks * 8;

    dlan_fused_kernel<<<blocks, 256, 0, stream>>>(x, W1, b1, W2, b2, WLd, bLd,
                                                  WLo, bLo, out, nrows, numTiles, nWaves);
}
